// MoELayer_71133248356528
// MI455X (gfx1250) — matrix operand order repaired and checked by reading, not compiled, not run on hardware
//
#include <hip/hip_runtime.h>

// ============================================================================
// MoE layer for MI455X (gfx1250, wave32, WMMA).
// All large GEMMs run on v_wmma_f32_16x16x32_bf16 (bf16 in, f32 accumulate).
// Fragment loaders follow the CDNA5 ISA 7.12.2 VGPR layouts:
//   A (16x32 bf16): lane m=l&15, k-half=(l>>4)*8; elems 0..7 -> k=kh+0..7,
//                   elems 8..15 -> k=16+kh+0..7 (two contiguous 16B chunks).
//   B (32x16 bf16): lane n=l&15, k-half=(l>>4)*8; elems 0..7 -> k=kh+0..7,
//                   elems 8..15 -> k=16+kh+0..7 (row-major [n,k] source).
//   C/D (16x16 f32): lane col n=l&15, row base mo=(l>>4)*8, VGPR r -> row mo+r.
// Round-2 changes:
//   * register blocking: each wave computes a 16x64 output strip (A fragment
//     reused 4x; 8x in the dual-weight SwiGLU kernel) -> ~4x higher FLOP/byte
//     so the GEMMs stay matrix-core bound instead of L2-bound.
//   * global_prefetch_b8 via __builtin_prefetch ahead of the K cursor.
//   * Tensor Data Mover (tensor_load_to_lds + s_wait_tensorcnt) stages the
//     per-token h0 row into LDS in the expert kernel.
// ============================================================================

#define DEVINL __device__ __forceinline__

typedef __attribute__((ext_vector_type(16))) __bf16 v16bf;
typedef __attribute__((ext_vector_type(8)))  float  v8f;
typedef __attribute__((ext_vector_type(4)))  unsigned int v4u;
typedef __attribute__((ext_vector_type(8)))  unsigned int v8u;

constexpr int Bb  = 4;
constexpr int Ss  = 2048;
constexpr int Dd  = 1024;
constexpr int Hh  = 2048;
constexpr int Aa  = 128;
constexpr int Nn  = Bb * Ss;   // 8192 tokens
constexpr int Gg  = 2;
constexpr int GSz = 4;

DEVINL v8f vzero() {
  v8f z;
#pragma unroll
  for (int i = 0; i < 8; ++i) z[i] = 0.f;
  return z;
}

DEVINL v8f wmma_bf16(v16bf a, v16bf b, v8f c) {
  return __builtin_amdgcn_wmma_f32_16x16x32_bf16(false, a, false, b, (short)0, c,
                                                 false, false);
}

// A fragment 16x32 from row-major [m, k] source with leading dim ld.
DEVINL v16bf frag_a_rm(const __bf16* src, int ld, int lane) {
  const int m  = lane & 15;
  const int kh = (lane >> 4) << 3;
  const __bf16* r = src + (size_t)m * ld + kh;
  v16bf a;
#pragma unroll
  for (int e = 0; e < 8; ++e) { a[e] = r[e]; a[e + 8] = r[16 + e]; }
  return a;
}

// B fragment 32x16 (k x n) from row-major [n, k] source with leading dim ld.
DEVINL v16bf frag_b_rm(const __bf16* src, int ld, int lane) {
  const int n  = lane & 15;
  const int kb = (lane >> 4) << 3;
  const __bf16* r = src + (size_t)n * ld + kb;
  v16bf b;
#pragma unroll
  for (int e = 0; e < 8; ++e) { b[e] = r[e]; b[e + 8] = r[16 + e]; }
  return b;
}

DEVINL float siluf_(float x) { return x / (1.f + __expf(-x)); }

DEVINL float wave_sum(float v) {
#pragma unroll
  for (int o = 16; o > 0; o >>= 1) v += __shfl_xor(v, o, 32);
  return v;
}

// ---------------------------------------------------------------------------
// Tensor Data Mover: stage one contiguous row of `nelem` f32 from global
// memory into LDS at byte offset lds_off. 2-SGPR-group descriptor form
// (tensors up to 2D, VADDR2/VADDR3 null). D# packing per CDNA5 ISA ch.8.
// ---------------------------------------------------------------------------
DEVINL void tdm_load_row_f32_to_lds(const float* gsrc, unsigned lds_off, int nelem) {
#if defined(__gfx1250__)
  unsigned long long ga = (unsigned long long)(size_t)gsrc;
  v4u g0;
  g0[0] = 1u;                                  // count=1, user descriptor
  g0[1] = lds_off;                             // lds_addr (bytes)
  g0[2] = (unsigned)ga;                        // global_addr[31:0]
  g0[3] = (unsigned)(ga >> 32) | (2u << 30);   // global_addr[56:32] | type=2
  v8u g1;
  g1[0] = 2u << 16;                            // wg_mask=0, data_size=4B
  g1[1] = (unsigned)nelem << 16;               // tensor_dim0 = nelem
  g1[2] = 1u << 16;                            // tensor_dim1 = 1
  g1[3] = (unsigned)nelem << 16;               // tile_dim0 = nelem
  g1[4] = 1u;                                  // tile_dim1 = 1, tile_dim2 = 0
  g1[5] = (unsigned)nelem;                     // tensor_dim0_stride low
  g1[6] = 0u;
  g1[7] = 0u;
  asm volatile("tensor_load_to_lds %0, %1" :: "s"(g0), "s"(g1) : "memory");
#if __has_builtin(__builtin_amdgcn_s_wait_tensorcnt)
  __builtin_amdgcn_s_wait_tensorcnt(0);
#else
  asm volatile("s_wait_tensorcnt 0" ::: "memory");
#endif
#else
  (void)gsrc; (void)lds_off; (void)nelem;
#endif
}

// ---------------------------------------------------------------------------
// Elementwise helpers
// ---------------------------------------------------------------------------
__global__ void k_f32_to_bf16(const float* in, __bf16* out, int n) {
  int i = blockIdx.x * blockDim.x + threadIdx.x;
  int stride = gridDim.x * blockDim.x;
  for (; i < n; i += stride) out[i] = (__bf16)in[i];
}

// in [R, C] f32 row-major -> out [C, R] bf16 row-major
__global__ void k_transpose_f32_to_bf16(const float* in, __bf16* out, int R, int C) {
  int i = blockIdx.x * blockDim.x + threadIdx.x;
  if (i < R * C) {
    int r = i / C, c = i % C;
    out[(size_t)c * R + r] = (__bf16)in[i];
  }
}

__global__ void k_zero(float* p, int n) {
  int i = blockIdx.x * blockDim.x + threadIdx.x;
  if (i < n) p[i] = 0.f;
}

// ---------------------------------------------------------------------------
// Router: per-token wave. gl = x.Wgr^T (G=2), el = x.Wer^T (GS=4),
// softmax, top-1 group, top-2 local experts, normalized weights, stats.
// ---------------------------------------------------------------------------
__global__ void __launch_bounds__(128)
k_router(const float* __restrict__ x, const float* __restrict__ Wgr,
         const float* __restrict__ Wer, int* idx, float* fw, float* wsum,
         float* stats) {
  const int lane = threadIdx.x & 31;
  const int tok  = blockIdx.x * 4 + (threadIdx.x >> 5);
  if (tok >= Nn) return;
  const float* xr = x + (size_t)tok * Dd;
  float g0 = 0, g1 = 0, e0 = 0, e1 = 0, e2 = 0, e3 = 0;
  for (int i = lane; i < Dd; i += 32) {
    float xv = xr[i];
    g0 += xv * Wgr[i];
    g1 += xv * Wgr[Dd + i];
    e0 += xv * Wer[i];
    e1 += xv * Wer[Dd + i];
    e2 += xv * Wer[2 * Dd + i];
    e3 += xv * Wer[3 * Dd + i];
  }
  g0 = wave_sum(g0); g1 = wave_sum(g1);
  e0 = wave_sum(e0); e1 = wave_sum(e1);
  e2 = wave_sum(e2); e3 = wave_sum(e3);
  if (lane == 0) {
    // group softmax + top-1 (ties -> lower index, matching lax.top_k)
    float mg = fmaxf(g0, g1);
    float p0 = __expf(g0 - mg), p1 = __expf(g1 - mg);
    float inv = 1.f / (p0 + p1);
    float gp0 = p0 * inv, gp1 = p1 * inv;
    int   gi = (gp1 > gp0) ? 1 : 0;
    float gw = (gp1 > gp0) ? gp1 : gp0;
    // expert softmax + top-2
    float el[4] = {e0, e1, e2, e3};
    float me = fmaxf(fmaxf(e0, e1), fmaxf(e2, e3));
    float ep[4]; float qs = 0.f;
#pragma unroll
    for (int i = 0; i < 4; ++i) { ep[i] = __expf(el[i] - me); qs += ep[i]; }
    float invq = 1.f / qs;
#pragma unroll
    for (int i = 0; i < 4; ++i) ep[i] *= invq;
    int i0 = 0; float b0 = ep[0];
    for (int i = 1; i < 4; ++i) if (ep[i] > b0) { b0 = ep[i]; i0 = i; }
    int i1 = (i0 == 0) ? 1 : 0; float b1 = ep[i1];
    for (int i = 0; i < 4; ++i)
      if (i != i0 && ep[i] > b1) { b1 = ep[i]; i1 = i; }
    float ls = 1.f / (b0 + b1 + 1e-7f);
    float f0 = gw * b0 * ls, f1 = gw * b1 * ls;
    int id0 = gi * GSz + i0, id1 = gi * GSz + i1;
    idx[2 * tok] = id0; idx[2 * tok + 1] = id1;
    fw[2 * tok] = f0;   fw[2 * tok + 1] = f1;
    wsum[tok] = f0 + f1;
    atomicAdd(&stats[id0], f0);
    atomicAdd(&stats[id1], f1);
    atomicAdd(&stats[8], g0 * g0 + g1 * g1);
    atomicAdd(&stats[9], e0 * e0 + e1 * e1 + e2 * e2 + e3 * e3);
  }
}

__global__ void k_rloss(const float* stats, float* rl) {
  if (threadIdx.x == 0 && blockIdx.x == 0) {
    float s = 0.f;
    for (int e = 0; e < 8; ++e) s += stats[e];
    float tgt = s / 8.f, mse = 0.f;
    for (int e = 0; e < 8; ++e) { float d = stats[e] - tgt; mse += d * d; }
    mse /= 8.f;
    float gl2 = stats[8] / ((float)Nn * (float)Gg);
    float el2 = stats[9] / ((float)Nn * (float)GSz);
    rl[0] = 0.001f * (mse + gl2 + el2);
  }
}

// ---------------------------------------------------------------------------
// Generic WMMA GEMMs: C[M,N] = A[M,K] * B[N,K]^T  (both bf16 sources row-major)
// One wave -> one 16x64 output strip (A fragment reused 4x). Nt % 64 == 0.
// ---------------------------------------------------------------------------
__global__ void __launch_bounds__(256)
k_gemm_f32(const __bf16* __restrict__ Am, int lda, const __bf16* __restrict__ Bm,
           int ldb, float* __restrict__ Cm, int ldc, int Mt, int Nt, int Kt) {
  const int wv = blockIdx.x * 8 + (threadIdx.x >> 5);
  const int nblk = Nt >> 6;
  if (wv >= (Mt >> 4) * nblk) return;
  const int tm = wv / nblk, tn = wv % nblk;
  const int lane = threadIdx.x & 31;
  const __bf16* ap = Am + (size_t)tm * 16 * lda;
  const __bf16* bp = Bm + (size_t)tn * 64 * ldb;
  v8f acc[4] = {vzero(), vzero(), vzero(), vzero()};
  for (int k = 0; k < Kt; k += 32) {
    __builtin_prefetch(ap + k + 64, 0, 0);
    v16bf a = frag_a_rm(ap + k, lda, lane);
#pragma unroll
    for (int j = 0; j < 4; ++j)
      acc[j] = wmma_bf16(a, frag_b_rm(bp + (size_t)j * 16 * ldb + k, ldb, lane),
                         acc[j]);
  }
  const int n = lane & 15, mo = (lane >> 4) << 3;
  float* o = Cm + (size_t)tm * 16 * ldc + tn * 64;
#pragma unroll
  for (int j = 0; j < 4; ++j)
#pragma unroll
    for (int r = 0; r < 8; ++r)
      o[(size_t)(mo + r) * ldc + j * 16 + n] = acc[j][r];
}

__global__ void __launch_bounds__(256)
k_gemm_bf16(const __bf16* __restrict__ Am, int lda, const __bf16* __restrict__ Bm,
            int ldb, __bf16* __restrict__ Cm, int ldc, int Mt, int Nt, int Kt) {
  const int wv = blockIdx.x * 8 + (threadIdx.x >> 5);
  const int nblk = Nt >> 6;
  if (wv >= (Mt >> 4) * nblk) return;
  const int tm = wv / nblk, tn = wv % nblk;
  const int lane = threadIdx.x & 31;
  const __bf16* ap = Am + (size_t)tm * 16 * lda;
  const __bf16* bp = Bm + (size_t)tn * 64 * ldb;
  v8f acc[4] = {vzero(), vzero(), vzero(), vzero()};
  for (int k = 0; k < Kt; k += 32) {
    v16bf a = frag_a_rm(ap + k, lda, lane);
#pragma unroll
    for (int j = 0; j < 4; ++j)
      acc[j] = wmma_bf16(a, frag_b_rm(bp + (size_t)j * 16 * ldb + k, ldb, lane),
                         acc[j]);
  }
  const int n = lane & 15, mo = (lane >> 4) << 3;
  __bf16* o = Cm + (size_t)tm * 16 * ldc + tn * 64;
#pragma unroll
  for (int j = 0; j < 4; ++j)
#pragma unroll
    for (int r = 0; r < 8; ++r)
      o[(size_t)(mo + r) * ldc + j * 16 + n] = (__bf16)acc[j][r];
}

// ---------------------------------------------------------------------------
// hidden = silu(x@Wg^T) * (x@Wu^T)  -> bf16 [N,H]
// Wave computes a 16x64 strip for BOTH weight matrices (A fragment reused 8x).
// ---------------------------------------------------------------------------
__global__ void __launch_bounds__(256)
k_hidden(const __bf16* __restrict__ xb, const __bf16* __restrict__ Wg,
         const __bf16* __restrict__ Wu, __bf16* __restrict__ hb) {
  const int wv = blockIdx.x * 8 + (threadIdx.x >> 5);
  const int nblk = Hh >> 6;                    // 32
  if (wv >= (Nn >> 4) * nblk) return;
  const int tm = wv / nblk, tn = wv % nblk;
  const int lane = threadIdx.x & 31;
  const __bf16* ap = xb + (size_t)tm * 16 * Dd;
  const __bf16* bg = Wg + (size_t)tn * 64 * Dd;
  const __bf16* bu = Wu + (size_t)tn * 64 * Dd;
  v8f ag[4], au[4];
#pragma unroll
  for (int j = 0; j < 4; ++j) { ag[j] = vzero(); au[j] = vzero(); }
  for (int k = 0; k < Dd; k += 32) {
    __builtin_prefetch(ap + k + 64, 0, 0);
    v16bf a = frag_a_rm(ap + k, Dd, lane);
#pragma unroll
    for (int j = 0; j < 4; ++j) {
      ag[j] = wmma_bf16(a, frag_b_rm(bg + (size_t)j * 16 * Dd + k, Dd, lane), ag[j]);
      au[j] = wmma_bf16(a, frag_b_rm(bu + (size_t)j * 16 * Dd + k, Dd, lane), au[j]);
    }
  }
  const int n = lane & 15, mo = (lane >> 4) << 3;
  __bf16* o = hb + (size_t)tm * 16 * Hh + tn * 64;
#pragma unroll
  for (int j = 0; j < 4; ++j)
#pragma unroll
    for (int r = 0; r < 8; ++r)
      o[(size_t)(mo + r) * Hh + j * 16 + n] = (__bf16)(siluf_(ag[j][r]) * au[j][r]);
}

// ---------------------------------------------------------------------------
// LayerNorm over A=128, wave per token. Optionally also scatters into a
// per-batch transposed copy [b][A][S] used as B-source by the attention GEMM.
// ---------------------------------------------------------------------------
__global__ void __launch_bounds__(128)
k_layernorm(const float* __restrict__ in, const float* __restrict__ g,
            const float* __restrict__ bta, __bf16* __restrict__ out,
            __bf16* outT, int ntok, int Sdim) {
  const int lane = threadIdx.x & 31;
  const int tok  = blockIdx.x * 4 + (threadIdx.x >> 5);
  if (tok >= ntok) return;
  const float* r = in + (size_t)tok * Aa;
  float v[4]; float s = 0.f;
#pragma unroll
  for (int j = 0; j < 4; ++j) { v[j] = r[lane + 32 * j]; s += v[j]; }
  s = wave_sum(s);
  const float mean = s * (1.f / Aa);
  float vs = 0.f;
#pragma unroll
  for (int j = 0; j < 4; ++j) { float d = v[j] - mean; vs += d * d; }
  vs = wave_sum(vs);
  const float rstd = rsqrtf(vs * (1.f / Aa) + 1e-5f);
#pragma unroll
  for (int j = 0; j < 4; ++j) {
    int c = lane + 32 * j;
    float o = (v[j] - mean) * rstd * g[c] + bta[c];
    out[(size_t)tok * Aa + c] = (__bf16)o;
    if (outT) {
      int bi = tok / Sdim, si = tok % Sdim;
      outT[((size_t)bi * Aa + c) * Sdim + si] = (__bf16)o;
    }
  }
}

// ---------------------------------------------------------------------------
// Fused adapter attention (per batch):
//   adapt[s,:] = sum_t silu(clip(ai[s]·ao[t])) * ai[t,:]
// Wave handles 16 s-rows, streams t in steps of 32. Score tiles round-trip
// through per-wave LDS to convert C-fragment layout -> A-fragment layout.
// ---------------------------------------------------------------------------
__global__ void __launch_bounds__(256)
k_attn(const __bf16* __restrict__ aib, const __bf16* __restrict__ aob,
       const __bf16* __restrict__ aiT, __bf16* __restrict__ adab) {
  __shared__ __bf16 scl[8][16 * 32];
  const int lane = threadIdx.x & 31;
  const int wid  = threadIdx.x >> 5;
  const int wv   = blockIdx.x * 8 + wid;
  const int sblocks = Ss / 16;
  if (wv >= Bb * sblocks) return;
  const int b  = wv / sblocks;
  const int sb = wv % sblocks;
  const __bf16* aiBase = aib + (size_t)b * Ss * Aa;   // [S, A]
  const __bf16* aoBase = aob + (size_t)b * Ss * Aa;   // [S, A]
  const __bf16* aiTB   = aiT + (size_t)b * Aa * Ss;   // [A, S]
  __bf16* my = scl[wid];
  const int n = lane & 15, mo = (lane >> 4) << 3;

  v16bf sa[4];
#pragma unroll
  for (int j = 0; j < 4; ++j)
    sa[j] = frag_a_rm(aiBase + (size_t)(sb * 16) * Aa + j * 32, Aa, lane);

  v8f acc[8];
#pragma unroll
  for (int j = 0; j < 8; ++j) acc[j] = vzero();

  for (int t = 0; t < Ss; t += 32) {
    v8f s0 = vzero(), s1 = vzero();
#pragma unroll
    for (int j = 0; j < 4; ++j) {
      v16bf b0 = frag_b_rm(aoBase + (size_t)t * Aa + j * 32, Aa, lane);
      v16bf b1 = frag_b_rm(aoBase + (size_t)(t + 16) * Aa + j * 32, Aa, lane);
      s0 = wmma_bf16(sa[j], b0, s0);
      s1 = wmma_bf16(sa[j], b1, s1);
    }
    // clip + silu, stage 16x32 score tile in LDS (row-major [m,k])
#pragma unroll
    for (int r = 0; r < 8; ++r) {
      float v0 = fminf(fmaxf(s0[r], -5.f), 5.f);
      float v1 = fminf(fmaxf(s1[r], -5.f), 5.f);
      my[(mo + r) * 32 + n]      = (__bf16)siluf_(v0);
      my[(mo + r) * 32 + 16 + n] = (__bf16)siluf_(v1);
    }
    asm volatile("s_wait_dscnt 0" ::: "memory");  // intra-wave LDS RAW
    v16bf af = frag_a_rm(my, 32, lane);           // scores as A (K = 32 t's)
    asm volatile("s_wait_dscnt 0" ::: "memory");  // reads done before rewrite
#pragma unroll
    for (int j = 0; j < 8; ++j) {
      v16bf bt = frag_b_rm(aiTB + (size_t)(j * 16) * Ss + t, Ss, lane);
      acc[j] = wmma_bf16(af, bt, acc[j]);
    }
  }
  __bf16* orow = adab + (size_t)(b * Ss + sb * 16) * Aa;
#pragma unroll
  for (int j = 0; j < 8; ++j)
#pragma unroll
    for (int r = 0; r < 8; ++r)
      orow[(size_t)(mo + r) * Aa + j * 16 + n] = (__bf16)acc[j][r];
}

// ---------------------------------------------------------------------------
// hidden += 0.1 * (adapt @ Wap^T)   (in-place on hb, K = A = 128), 16x64/wave
// ---------------------------------------------------------------------------
__global__ void __launch_bounds__(256)
k_hidden2(const __bf16* __restrict__ adab, const __bf16* __restrict__ Wap,
          __bf16* __restrict__ hb) {
  const int wv = blockIdx.x * 8 + (threadIdx.x >> 5);
  const int nblk = Hh >> 6;
  if (wv >= (Nn >> 4) * nblk) return;
  const int tm = wv / nblk, tn = wv % nblk;
  const int lane = threadIdx.x & 31;
  const __bf16* ap = adab + (size_t)tm * 16 * Aa;
  const __bf16* bp = Wap + (size_t)tn * 64 * Aa;
  v8f acc[4] = {vzero(), vzero(), vzero(), vzero()};
  for (int k = 0; k < Aa; k += 32) {
    v16bf a = frag_a_rm(ap + k, Aa, lane);
#pragma unroll
    for (int j = 0; j < 4; ++j)
      acc[j] = wmma_bf16(a, frag_b_rm(bp + (size_t)j * 16 * Aa + k, Aa, lane),
                         acc[j]);
  }
  const int n = lane & 15, mo = (lane >> 4) << 3;
  __bf16* o = hb + (size_t)tm * 16 * Hh + tn * 64;
#pragma unroll
  for (int j = 0; j < 4; ++j)
#pragma unroll
    for (int r = 0; r < 8; ++r) {
      size_t p = (size_t)(mo + r) * Hh + j * 16 + n;
      o[p] = (__bf16)((float)o[p] + 0.1f * acc[j][r]);
    }
}

// ---------------------------------------------------------------------------
// Sparse expert path: per token, only the 2 routed experts.
// hew[n,c] = sum_k w_k * LN(h0[n]@Wea[e_k]^T ; ln_eg[e_k], ln_eb[e_k])[c]
// The h0 row is staged into LDS by the Tensor Data Mover (TDM).
// ---------------------------------------------------------------------------
__global__ void __launch_bounds__(128)
k_expert(const float* __restrict__ h0, const float* __restrict__ Wea,
         const float* __restrict__ eg, const float* __restrict__ eb,
         const int* __restrict__ idx, const float* __restrict__ fw,
         __bf16* __restrict__ hewb) {
  __shared__ float hrow[4][Aa];
  const int lane = threadIdx.x & 31;
  const int wid  = threadIdx.x >> 5;
  const int tok  = blockIdx.x * 4 + wid;
  if (tok >= Nn) return;
  float* hr = hrow[wid];
  const float* h = h0 + (size_t)tok * Aa;
#if defined(__gfx1250__)
  // hrow is the only LDS object -> its segment offset is 0; wave wid's row
  // lives at byte offset wid*Aa*4.
  tdm_load_row_f32_to_lds(h, (unsigned)(wid * Aa * 4), Aa);
#else
#pragma unroll
  for (int j = 0; j < 4; ++j) hr[lane + 32 * j] = h[lane + 32 * j];
  asm volatile("s_wait_dscnt 0" ::: "memory");
#endif
  float res[4] = {0.f, 0.f, 0.f, 0.f};
  for (int kk = 0; kk < 2; ++kk) {
    const int   e = idx[2 * tok + kk];
    const float w = fw[2 * tok + kk];
    const float* We = Wea + (size_t)e * Aa * Aa;
    float t[4];
#pragma unroll
    for (int j = 0; j < 4; ++j) {
      const int c = lane + 32 * j;
      const float* wr = We + (size_t)c * Aa;
      float s = 0.f;
      for (int a = 0; a < Aa; ++a) s += hr[a] * wr[a];
      t[j] = s;
    }
    float s = wave_sum(t[0] + t[1] + t[2] + t[3]);
    const float mean = s * (1.f / Aa);
    float vs = 0.f;
#pragma unroll
    for (int j = 0; j < 4; ++j) { float d = t[j] - mean; vs += d * d; }
    vs = wave_sum(vs);
    const float rstd = rsqrtf(vs * (1.f / Aa) + 1e-5f);
#pragma unroll
    for (int j = 0; j < 4; ++j) {
      const int c = lane + 32 * j;
      res[j] += w * ((t[j] - mean) * rstd * eg[e * Aa + c] + eb[e * Aa + c]);
    }
  }
#pragma unroll
  for (int j = 0; j < 4; ++j)
    hewb[(size_t)tok * Aa + lane + 32 * j] = (__bf16)res[j];
}

// ---------------------------------------------------------------------------
// Final: out = wsum[row] * (h2 @ Wd^T) + 0.1 * (hew @ M^T)
// 16x64 strip per wave; two WMMA K-loops (K=2048 and K=128) per strip.
// ---------------------------------------------------------------------------
__global__ void __launch_bounds__(256)
k_final(const __bf16* __restrict__ h2, const __bf16* __restrict__ Wd,
        const __bf16* __restrict__ hew, const __bf16* __restrict__ Mb,
        const float* __restrict__ wsum, float* __restrict__ out) {
  const int wv = blockIdx.x * 8 + (threadIdx.x >> 5);
  const int nblk = Dd >> 6;                    // 16
  if (wv >= (Nn >> 4) * nblk) return;
  const int tm = wv / nblk, tn = wv % nblk;
  const int lane = threadIdx.x & 31;
  v8f a1[4] = {vzero(), vzero(), vzero(), vzero()};
  v8f a2[4] = {vzero(), vzero(), vzero(), vzero()};
  {
    const __bf16* ap = h2 + (size_t)tm * 16 * Hh;
    const __bf16* bp = Wd + (size_t)tn * 64 * Hh;
    for (int k = 0; k < Hh; k += 32) {
      __builtin_prefetch(ap + k + 64, 0, 0);
      v16bf a = frag_a_rm(ap + k, Hh, lane);
#pragma unroll
      for (int j = 0; j < 4; ++j)
        a1[j] = wmma_bf16(a, frag_b_rm(bp + (size_t)j * 16 * Hh + k, Hh, lane),
                          a1[j]);
    }
  }
  {
    const __bf16* ap = hew + (size_t)tm * 16 * Aa;
    const __bf16* bp = Mb + (size_t)tn * 64 * Aa;
    for (int k = 0; k < Aa; k += 32) {
      v16bf a = frag_a_rm(ap + k, Aa, lane);
#pragma unroll
      for (int j = 0; j < 4; ++j)
        a2[j] = wmma_bf16(a, frag_b_rm(bp + (size_t)j * 16 * Aa + k, Aa, lane),
                          a2[j]);
    }
  }
  const int n = lane & 15, mo = (lane >> 4) << 3;
#pragma unroll
  for (int j = 0; j < 4; ++j)
#pragma unroll
    for (int r = 0; r < 8; ++r) {
      const int row = tm * 16 + mo + r;
      out[(size_t)row * Dd + tn * 64 + j * 16 + n] =
          wsum[row] * a1[j][r] + 0.1f * a2[j][r];
    }
}

// ===========================================================================
// Host orchestration
// ===========================================================================
extern "C" void kernel_launch(void* const* d_in, const int* in_sizes, int n_in,
                              void* d_out, int out_size, void* d_ws, size_t ws_size,
                              hipStream_t stream) {
  const float* x     = (const float*)d_in[0];
  const float* Wu    = (const float*)d_in[1];
  const float* Wg    = (const float*)d_in[2];
  const float* Wd    = (const float*)d_in[3];
  const float* Wpre  = (const float*)d_in[4];
  const float* Wpost = (const float*)d_in[5];
  const float* ln_g  = (const float*)d_in[6];
  const float* ln_b  = (const float*)d_in[7];
  const float* Wap   = (const float*)d_in[8];
  const float* Wea   = (const float*)d_in[9];
  const float* ln_eg = (const float*)d_in[10];
  const float* ln_eb = (const float*)d_in[11];
  const float* Wep   = (const float*)d_in[12];
  const float* Wop   = (const float*)d_in[13];
  const float* Wgr   = (const float*)d_in[14];
  const float* Wer   = (const float*)d_in[15];
  float* out = (float*)d_out;

  // Workspace carve-up (~88 MB), 256B aligned slots.
  char* w = (char*)d_ws;
  size_t off = 0;
  auto take = [&](size_t bytes) -> void* {
    void* p = w + off;
    off += (bytes + 255) & ~(size_t)255;
    return p;
  };
  __bf16* xb      = (__bf16*)take((size_t)Nn * Dd * 2);
  __bf16* Wg_b    = (__bf16*)take((size_t)Hh * Dd * 2);
  __bf16* Wu_b    = (__bf16*)take((size_t)Hh * Dd * 2);
  __bf16* Wd_b    = (__bf16*)take((size_t)Dd * Hh * 2);
  __bf16* Wop_b   = (__bf16*)take((size_t)Dd * Hh * 2);
  __bf16* Wpre_b  = (__bf16*)take((size_t)Aa * Dd * 2);
  __bf16* Wpost_b = (__bf16*)take((size_t)Aa * Hh * 2);
  __bf16* Wap_b   = (__bf16*)take((size_t)Hh * Aa * 2);
  __bf16* WepT    = (__bf16*)take((size_t)Aa * Hh * 2);   // Wep^T [A,H]
  __bf16* Mb      = (__bf16*)take((size_t)Dd * Aa * 2);   // Wop@Wep [D,A]
  __bf16* hb      = (__bf16*)take((size_t)Nn * Hh * 2);   // hidden (later h2, in place)
  float*  h0      = (float*) take((size_t)Nn * Aa * 4);   // x@Wpre^T (pre-LN)
  float*  aopre   = (float*) take((size_t)Nn * Aa * 4);   // hidden@Wpost^T (pre-LN)
  __bf16* aib     = (__bf16*)take((size_t)Nn * Aa * 2);   // LN(h0)
  __bf16* aiT     = (__bf16*)take((size_t)Bb * Aa * Ss * 2); // per-batch ai^T
  __bf16* aob     = (__bf16*)take((size_t)Nn * Aa * 2);   // LN(aopre)
  __bf16* adab    = (__bf16*)take((size_t)Nn * Aa * 2);   // adapter attn output
  __bf16* hewb    = (__bf16*)take((size_t)Nn * Aa * 2);   // weighted expert output
  float*  wsum    = (float*) take((size_t)Nn * 4);
  int*    idxbuf  = (int*)   take((size_t)Nn * 2 * 4);
  float*  fwbuf   = (float*) take((size_t)Nn * 2 * 4);
  float*  stats   = (float*) take(16 * 4);
  (void)in_sizes; (void)n_in; (void)out_size; (void)ws_size;

  // --- stage 0: init + precision conversion ---
  k_zero<<<1, 32, 0, stream>>>(stats, 16);
  k_f32_to_bf16<<<4096, 256, 0, stream>>>(x, xb, Nn * Dd);
  k_f32_to_bf16<<<2048, 256, 0, stream>>>(Wg, Wg_b, Hh * Dd);
  k_f32_to_bf16<<<2048, 256, 0, stream>>>(Wu, Wu_b, Hh * Dd);
  k_f32_to_bf16<<<2048, 256, 0, stream>>>(Wd, Wd_b, Dd * Hh);
  k_f32_to_bf16<<<2048, 256, 0, stream>>>(Wop, Wop_b, Dd * Hh);
  k_f32_to_bf16<<<512, 256, 0, stream>>>(Wpre, Wpre_b, Aa * Dd);
  k_f32_to_bf16<<<512, 256, 0, stream>>>(Wpost, Wpost_b, Aa * Hh);
  k_f32_to_bf16<<<512, 256, 0, stream>>>(Wap, Wap_b, Hh * Aa);
  k_transpose_f32_to_bf16<<<(Hh * Aa + 255) / 256, 256, 0, stream>>>(Wep, WepT, Hh, Aa);

  // --- stage 1: router + rloss ---
  k_router<<<Nn / 4, 128, 0, stream>>>(x, Wgr, Wer, idxbuf, fwbuf, wsum, stats);
  k_rloss<<<1, 32, 0, stream>>>(stats, out + (size_t)Nn * Dd);

  // --- stage 2: M = Wop @ Wep  [D,A] ---
  k_gemm_bf16<<<16, 256, 0, stream>>>(Wop_b, Hh, WepT, Hh, Mb, Aa, Dd, Aa, Hh);

  // --- stage 3: hidden = silu(x@Wg^T)*(x@Wu^T) ---
  k_hidden<<<2048, 256, 0, stream>>>(xb, Wg_b, Wu_b, hb);

  // --- stage 4: h0 = x@Wpre^T ; ai = LN(h0) (+ transposed copy) ---
  k_gemm_f32<<<128, 256, 0, stream>>>(xb, Dd, Wpre_b, Dd, h0, Aa, Nn, Aa, Dd);
  k_layernorm<<<Nn / 4, 128, 0, stream>>>(h0, ln_g, ln_b, aib, aiT, Nn, Ss);

  // --- stage 5: ao = LN(hidden@Wpost^T) ---
  k_gemm_f32<<<128, 256, 0, stream>>>(hb, Hh, Wpost_b, Hh, aopre, Aa, Nn, Aa, Hh);
  k_layernorm<<<Nn / 4, 128, 0, stream>>>(aopre, ln_g, ln_b, aob, nullptr, Nn, Ss);

  // --- stage 6: fused adapter attention ---
  k_attn<<<64, 256, 0, stream>>>(aib, aob, aiT, adab);

  // --- stage 7: hidden += 0.1 * adapt@Wap^T (in place) ---
  k_hidden2<<<2048, 256, 0, stream>>>(adab, Wap_b, hb);

  // --- stage 8: sparse expert path (TDM-staged h0 rows) ---
  k_expert<<<Nn / 4, 128, 0, stream>>>(h0, Wea, ln_eg, ln_eb, idxbuf, fwbuf, hewb);

  // --- stage 9: out = wsum*(h2@Wd^T) + 0.1*(hew@M^T) ---
  k_final<<<1024, 256, 0, stream>>>(hb, Wd_b, hewb, Mb, wsum, out);
}
